// AllAtomGCN_85624468013347
// MI455X (gfx1250) — matrix operand order repaired and checked by reading, not compiled, not run on hardware
//
#include <hip/hip_runtime.h>
#include <hip/hip_bf16.h>

typedef __attribute__((ext_vector_type(16))) _Float16 v16h;
typedef __attribute__((ext_vector_type(8)))  _Float16 v8h;
typedef __attribute__((ext_vector_type(8)))  float    v8f;

#define H 32
#define IN 64
#define EPS 1e-5f

__device__ __forceinline__ float eluf(float x) {
    return x > 0.0f ? x : (__expf(x) - 1.0f);
}

// LayerNorm + ELU over one 32-float row (in LDS), store as f16 row (4x16B).
__device__ __forceinline__ void ln_elu_store(const float* row,
                                             const float* __restrict__ g,
                                             const float* __restrict__ be,
                                             v8h* __restrict__ out) {
    float x[H];
    const float4* r4 = (const float4*)row;
#pragma unroll
    for (int q = 0; q < 8; ++q) {
        float4 v = r4[q];
        x[q * 4 + 0] = v.x; x[q * 4 + 1] = v.y;
        x[q * 4 + 2] = v.z; x[q * 4 + 3] = v.w;
    }
    float mu = 0.0f;
#pragma unroll
    for (int j = 0; j < H; ++j) mu += x[j];
    mu *= (1.0f / H);
    float var = 0.0f;
#pragma unroll
    for (int j = 0; j < H; ++j) { float d = x[j] - mu; var += d * d; }
    var *= (1.0f / H);
    float rs = rsqrtf(var + EPS);
    v8h o[4];
#pragma unroll
    for (int j = 0; j < H; ++j) {
        float y = (x[j] - mu) * rs * g[j] + be[j];
        o[j >> 3][j & 7] = (_Float16)eluf(y);
    }
#pragma unroll
    for (int q = 0; q < 4; ++q) out[q] = o[q];
}

// ---------------- utility kernels ----------------

__global__ void k_zero(float* __restrict__ p, size_t n) {
    size_t t = (size_t)blockIdx.x * blockDim.x + threadIdx.x;
    if (t < n) p[t] = 0.0f;
}

__global__ void k_deg(const int* __restrict__ src, const int* __restrict__ dst,
                      float* __restrict__ degs, float* __restrict__ degd, int E) {
    int t = blockIdx.x * blockDim.x + threadIdx.x;
    if (t < E) {
        atomicAdd(&degs[src[t]], 1.0f);
        atomicAdd(&degd[dst[t]], 1.0f);
    }
}

__global__ void k_norm(float* __restrict__ ns, float* __restrict__ nd, int N) {
    int t = blockIdx.x * blockDim.x + threadIdx.x;
    if (t < N) {
        float a = ns[t];
        ns[t] = a > 0.0f ? rsqrtf(a) : 0.0f;
        float b = nd[t];
        nd[t] = b > 0.0f ? rsqrtf(b) : 0.0f;
    }
}

// ---------------- input MLP: h16 = f16(ELU(LN(feats @ w0 + b0))) ----------------
// One wave32 per 16-node tile. Weights staged in LDS pre-swizzled into WMMA
// B-fragment order so each lane's fragment is one contiguous 32B LDS load.

__global__ __launch_bounds__(256) void k_in_mlp(
    const float* __restrict__ feats, const float* __restrict__ w0,
    const float* __restrict__ b0, const float* __restrict__ g0,
    const float* __restrict__ be0, v8h* __restrict__ h16, int ntiles)
{
    // fragment order: [ks][nt][lane][j], j contiguous
    __shared__ __align__(32) _Float16 sWf[2 * 2 * 32 * 16];   // 4KB
    __shared__ float sT[8][16 * H];                           // 16KB

    for (int idx = threadIdx.x; idx < 2048; idx += 256) {
        int j    = idx & 15;
        int lane = (idx >> 4) & 31;
        int nt   = (idx >> 9) & 1;
        int ks   = idx >> 10;
        int k = ks * 32 + ((lane >> 4) ? 8 : 0) + (j & 7) + ((j >> 3) ? 16 : 0);
        int n = nt * 16 + (lane & 15);
        sWf[idx] = (_Float16)w0[k * H + n];
    }
    __syncthreads();

    int wave = threadIdx.x >> 5;
    int lane = threadIdx.x & 31;
    int tile = blockIdx.x * 8 + wave;
    if (tile >= ntiles) return;            // wave-uniform: EXEC all-1 inside

    int base  = tile * 16;
    int m     = lane & 15;
    int hi    = lane >> 4;
    int kbA   = hi * 8;
    int ncol0 = lane & 15;

    const float* arow = feats + (size_t)(base + m) * IN;

    v8f acc[2];
#pragma unroll
    for (int nt = 0; nt < 2; ++nt) {
        float bias = b0[ncol0 + nt * 16];
#pragma unroll
        for (int r = 0; r < 8; ++r) acc[nt][r] = bias;
    }

#pragma unroll
    for (int ks = 0; ks < 2; ++ks) {
        v16h a;
#pragma unroll
        for (int j = 0; j < 8; ++j) {
            a[j]     = (_Float16)arow[ks * 32 + kbA + j];
            a[8 + j] = (_Float16)arow[ks * 32 + kbA + 16 + j];
        }
#pragma unroll
        for (int nt = 0; nt < 2; ++nt) {
            v16h b = *(const v16h*)&sWf[((ks * 2 + nt) * 32 + lane) * 16];
            acc[nt] = __builtin_amdgcn_wmma_f32_16x16x32_f16(
                false, a, false, b, (short)0, acc[nt], false, false);
        }
    }

#pragma unroll
    for (int nt = 0; nt < 2; ++nt) {
        int ncol = ncol0 + nt * 16;
#pragma unroll
        for (int r = 0; r < 8; ++r)
            sT[wave][(r + hi * 8) * H + ncol] = acc[nt][r];
    }

    if (lane < 16)
        ln_elu_store(&sT[wave][lane * H], g0, be0, h16 + (size_t)(base + lane) * 4);
}

// ---------------- edge gather/scatter: macc[dst] += h16[src] * ns[src] ----------------
// 4 threads per edge, 8 f16 (16B) per thread; f32 accumulator in L2.

__global__ void k_gather(const v8h* __restrict__ h8, const int* __restrict__ src,
                         const int* __restrict__ dst, const float* __restrict__ ns,
                         float* __restrict__ macc, int E4)
{
    int t = blockIdx.x * blockDim.x + threadIdx.x;
    if (t >= E4) return;
    int e = t >> 2, c = t & 3;
    int s = src[e], d = dst[e];
    float sc = ns[s];
    v8h v = h8[(size_t)s * 4 + c];
    float* mp = macc + (size_t)d * H + c * 8;
#pragma unroll
    for (int j = 0; j < 8; ++j)
        atomicAdd(mp + j, (float)v[j] * sc);
}

// ---------------- fused block: h16 = f16(ELU(LN(ELU((macc*nd) @ wc + bc) @ w + b))) ----------------

__global__ __launch_bounds__(256) void k_block(
    const float* __restrict__ macc, const float* __restrict__ nd,
    const float* __restrict__ wc, const float* __restrict__ bc,
    const float* __restrict__ w,  const float* __restrict__ b,
    const float* __restrict__ g,  const float* __restrict__ be,
    v8h* __restrict__ h16, int ntiles)
{
    // fragment order: [nt][lane][j]
    __shared__ __align__(32) _Float16 sWcF[2 * 32 * 16];   // 2KB
    __shared__ __align__(32) _Float16 sW2F[2 * 32 * 16];   // 2KB
    __shared__ float sT[8][16 * H];                        // 16KB

    for (int idx = threadIdx.x; idx < 1024; idx += 256) {
        int j    = idx & 15;
        int lane = (idx >> 4) & 31;
        int nt   = idx >> 9;
        int k = ((lane >> 4) ? 8 : 0) + (j & 7) + ((j >> 3) ? 16 : 0);
        int n = nt * 16 + (lane & 15);
        sWcF[idx] = (_Float16)wc[k * H + n];
        sW2F[idx] = (_Float16)w[k * H + n];
    }
    __syncthreads();

    int wave = threadIdx.x >> 5;
    int lane = threadIdx.x & 31;
    int tile = blockIdx.x * 8 + wave;
    if (tile >= ntiles) return;

    int base  = tile * 16;
    int m     = lane & 15;
    int hi    = lane >> 4;
    int kbA   = hi * 8;
    int ncol0 = lane & 15;

    // A fragment = row of (macc * nd), K=32 single step
    const float* arow = macc + (size_t)(base + m) * H;
    float ndv = nd[base + m];
    v16h a;
#pragma unroll
    for (int j = 0; j < 8; ++j) {
        a[j]     = (_Float16)(arow[kbA + j] * ndv);
        a[8 + j] = (_Float16)(arow[kbA + 16 + j] * ndv);
    }

    // GEMM1: t1 = ELU(m @ wc + bc), staged to LDS for A-reshape
    v8f acc[2];
#pragma unroll
    for (int nt = 0; nt < 2; ++nt) {
        int ncol = ncol0 + nt * 16;
        float bias = bc[ncol];
#pragma unroll
        for (int r = 0; r < 8; ++r) acc[nt][r] = bias;
        v16h bb = *(const v16h*)&sWcF[(nt * 32 + lane) * 16];
        acc[nt] = __builtin_amdgcn_wmma_f32_16x16x32_f16(
            false, a, false, bb, (short)0, acc[nt], false, false);
#pragma unroll
        for (int r = 0; r < 8; ++r) acc[nt][r] = eluf(acc[nt][r]);
#pragma unroll
        for (int r = 0; r < 8; ++r)
            sT[wave][(r + hi * 8) * H + ncol] = acc[nt][r];
    }

    // reload t1 row as A fragment
    const float* trow = &sT[wave][m * H];
    v16h a2;
#pragma unroll
    for (int j = 0; j < 8; ++j) {
        a2[j]     = (_Float16)trow[kbA + j];
        a2[8 + j] = (_Float16)trow[kbA + 16 + j];
    }

    // GEMM2: t2 = t1 @ w + b
    v8f acc2[2];
#pragma unroll
    for (int nt = 0; nt < 2; ++nt) {
        int ncol = ncol0 + nt * 16;
        float bias = b[ncol];
#pragma unroll
        for (int r = 0; r < 8; ++r) acc2[nt][r] = bias;
        v16h bb = *(const v16h*)&sW2F[(nt * 32 + lane) * 16];
        acc2[nt] = __builtin_amdgcn_wmma_f32_16x16x32_f16(
            false, a2, false, bb, (short)0, acc2[nt], false, false);
#pragma unroll
        for (int r = 0; r < 8; ++r)
            sT[wave][(r + hi * 8) * H + ncol] = acc2[nt][r];
    }

    if (lane < 16)
        ln_elu_store(&sT[wave][lane * H], g, be, h16 + (size_t)(base + lane) * 4);
}

// ---------------- output head + per-graph readout ----------------

__global__ void k_head(const v8h* __restrict__ h16, const float* __restrict__ wout,
                       const float* __restrict__ bout, const int* __restrict__ gid,
                       float* __restrict__ out, int N)
{
    int t = blockIdx.x * blockDim.x + threadIdx.x;
    if (t < N) {
        float acc = bout[0];
#pragma unroll
        for (int q = 0; q < 4; ++q) {
            v8h v = h16[(size_t)t * 4 + q];
#pragma unroll
            for (int j = 0; j < 8; ++j)
                acc += (float)v[j] * wout[q * 8 + j];
        }
        acc = eluf(acc);
        atomicAdd(&out[gid[t]], acc);
    }
}

// ---------------- launch ----------------

extern "C" void kernel_launch(void* const* d_in, const int* in_sizes, int n_in,
                              void* d_out, int out_size, void* d_ws, size_t ws_size,
                              hipStream_t stream) {
    const float* feats = (const float*)d_in[0];
    const int*   src   = (const int*)d_in[1];
    const int*   dst   = (const int*)d_in[2];
    const int*   gid   = (const int*)d_in[3];
    const float* w0  = (const float*)d_in[4];
    const float* b0  = (const float*)d_in[5];
    const float* g0  = (const float*)d_in[6];
    const float* be0 = (const float*)d_in[7];
    const float* wc1 = (const float*)d_in[8];
    const float* bc1 = (const float*)d_in[9];
    const float* w1  = (const float*)d_in[10];
    const float* b1  = (const float*)d_in[11];
    const float* g1  = (const float*)d_in[12];
    const float* be1 = (const float*)d_in[13];
    const float* wc2 = (const float*)d_in[14];
    const float* bc2 = (const float*)d_in[15];
    const float* w2  = (const float*)d_in[16];
    const float* b2  = (const float*)d_in[17];
    const float* g2  = (const float*)d_in[18];
    const float* be2 = (const float*)d_in[19];
    const float* wout = (const float*)d_in[20];
    const float* bout = (const float*)d_in[21];
    float* out = (float*)d_out;

    const int N = in_sizes[3];       // 1,000,000
    const int E = in_sizes[1];       // 4,000,000
    const int ntiles = N / 16;       // exact: 62,500

    // workspace: ns[N] f32 | nd[N] f32 | h16[32N] f16 | macc[32N] f32  ~= 200 MB
    float* ns   = (float*)d_ws;
    float* nd   = ns + N;
    v8h*   h16  = (v8h*)(nd + N);                       // N*4 v8h = 32N f16
    float* macc = (float*)((_Float16*)(nd + N) + (size_t)N * H);

    const int TB = 256;
    dim3 blk(TB);
    auto gsz = [&](size_t n) { return dim3((unsigned)((n + TB - 1) / TB)); };

    // degrees -> symmetric norms
    k_zero<<<gsz((size_t)2 * N), blk, 0, stream>>>(ns, (size_t)2 * N);
    k_deg<<<gsz(E), blk, 0, stream>>>(src, dst, ns, nd, E);
    k_norm<<<gsz(N), blk, 0, stream>>>(ns, nd, N);

    // input MLP
    k_in_mlp<<<dim3((ntiles + 7) / 8), blk, 0, stream>>>(feats, w0, b0, g0, be0, h16, ntiles);

    // block 1
    k_zero<<<gsz((size_t)N * H), blk, 0, stream>>>(macc, (size_t)N * H);
    k_gather<<<gsz((size_t)E * 4), blk, 0, stream>>>(h16, src, dst, ns, macc, E * 4);
    k_block<<<dim3((ntiles + 7) / 8), blk, 0, stream>>>(macc, nd, wc1, bc1, w1, b1, g1, be1, h16, ntiles);

    // block 2
    k_zero<<<gsz((size_t)N * H), blk, 0, stream>>>(macc, (size_t)N * H);
    k_gather<<<gsz((size_t)E * 4), blk, 0, stream>>>(h16, src, dst, ns, macc, E * 4);
    k_block<<<dim3((ntiles + 7) / 8), blk, 0, stream>>>(macc, nd, wc2, bc2, w2, b2, g2, be2, h16, ntiles);

    // head + readout
    k_zero<<<gsz(out_size), blk, 0, stream>>>(out, (size_t)out_size);
    k_head<<<gsz(N), blk, 0, stream>>>(h16, wout, bout, gid, out, N);

    (void)n_in; (void)ws_size;
}
